// MoE_42984032698463
// MI455X (gfx1250) — matrix operand order repaired and checked by reading, not compiled, not run on hardware
//
#include <hip/hip_runtime.h>

typedef _Float16 h16;
typedef __attribute__((ext_vector_type(16))) _Float16 v16h;
typedef __attribute__((ext_vector_type(8)))  float    v8f;

#define EXPERTS 4
#define CCH 128
#define BB 4
#define LL 1024          // H*W
#define DIN 256
#define DSTATE 16
#define DTRANK 8
#define DCONV 4
#define MROWS (BB*LL)    // 4096

// ---------------------------------------------------------------------------
// Generic f16 WMMA GEMM:  out[e] = A[e] (MxK, row major) @ W[e]^T (W is NxK)
// One wave = one 16x16 tile. 8 waves per block over N tiles.
// GUARD=true only for the N=40 (padded to 48) x_proj GEMM; the other GEMMs
// get a branch-free inner loop (no EXEC save/restore, no b-zeroing).
// ---------------------------------------------------------------------------
template<bool GUARD>
__global__ __launch_bounds__(256)
void k_gemm_f16(const h16* __restrict__ A, int lda, long strideAe,
                const h16* __restrict__ W, int ldw, long strideWe,
                const float* __restrict__ bias, int strideBiasE,
                void* __restrict__ out, int ldo, long strideOe, int outHalf,
                int K, int Nvalid, int tilesN)
{
    const int e    = blockIdx.z;
    const int wv   = threadIdx.x >> 5;
    const int lane = threadIdx.x & 31;
    const int ntile = blockIdx.x * 8 + wv;
    if (ntile >= tilesN) return;                 // wave-uniform guard
    const int mtile = blockIdx.y;
    const int lr = lane & 15, hi = lane >> 4;

    const h16* Ae = A + (size_t)e * strideAe;
    const h16* We = W + (size_t)e * strideWe;

    const int row = mtile * 16 + lr;             // A row for this lane
    const int col = ntile * 16 + lr;             // B column for this lane

    v8f acc;
    #pragma unroll
    for (int i = 0; i < 8; ++i) acc[i] = 0.0f;

    const h16* ap = Ae + (size_t)row * lda + hi * 16;
    const h16* bp = We + (size_t)col * ldw + hi * 16;

    for (int k0 = 0; k0 < K; k0 += 32) {
        // A 16x32 f16 layout: lane<16 -> K k0..k0+15 ; lane>=16 -> K k0+16..k0+31
        v16h a;
        #pragma unroll
        for (int j = 0; j < 8; ++j) a[j]     = ap[k0 + j];
        #pragma unroll
        for (int j = 0; j < 8; ++j) a[8 + j] = ap[k0 + 8 + j];

        // B 32x16 f16 layout: lane<16 -> col=lr, K k0..k0+15 ; lane>=16 -> K k0+16..k0+31
        v16h b;
        if (GUARD) {
            #pragma unroll
            for (int j = 0; j < 16; ++j) b[j] = (_Float16)0.0f;
            if (col < Nvalid) {
                #pragma unroll
                for (int j = 0; j < 16; ++j) b[j] = bp[k0 + j];
            }
        } else {
            #pragma unroll
            for (int j = 0; j < 16; ++j) b[j] = bp[k0 + j];
        }
        acc = __builtin_amdgcn_wmma_f32_16x16x32_f16(false, a, false, b,
                                                     (short)0, acc, false, false);
    }

    float bv = 0.0f;
    if (bias && (!GUARD || col < Nvalid)) bv = bias[(size_t)e * strideBiasE + col];

    #pragma unroll
    for (int i = 0; i < 8; ++i) {
        const int orow = mtile * 16 + hi * 8 + i;    // C/D layout
        const float v = acc[i] + bv;
        const size_t oidx = (size_t)e * strideOe + (size_t)orow * ldo + col;
        if (outHalf) ((h16*)out)[oidx] = (h16)v;
        else         ((float*)out)[oidx] = v;
    }
}

// ---------------------------------------------------------------------------
// LayerNorm over C=128 of x (B,C,H,W) -> xhat (B*L, C) in f16 (affine folded
// into in_proj weights elsewhere).  One block per (b,l), 128 threads.
// ---------------------------------------------------------------------------
__global__ __launch_bounds__(128)
void k_ln(const float* __restrict__ x, h16* __restrict__ xhat)
{
    const int bl = blockIdx.x;
    const int b  = bl >> 10;           // L = 1024
    const int l  = bl & 1023;
    const int c  = threadIdx.x;
    const float v = x[((size_t)b * CCH + c) * LL + l];
    __shared__ float red[128];
    red[c] = v; __syncthreads();
    for (int s = 64; s > 0; s >>= 1) { if (c < s) red[c] += red[c + s]; __syncthreads(); }
    const float mu = red[0] * (1.0f / 128.0f);
    __syncthreads();
    const float d = v - mu;
    red[c] = d * d; __syncthreads();
    for (int s = 64; s > 0; s >>= 1) { if (c < s) red[c] += red[c + s]; __syncthreads(); }
    const float var = red[0] * (1.0f / 128.0f);
    xhat[(size_t)bl * CCH + c] = (h16)(d * rsqrtf(var + 1e-5f));
}

// Global average pool: x_gap[b][c] = mean over H*W. One block per (b,c).
__global__ __launch_bounds__(256)
void k_gap(const float* __restrict__ x, float* __restrict__ xgap)
{
    const int bc = blockIdx.x;
    const int t  = threadIdx.x;
    const float* p = x + (size_t)bc * LL;
    float s = 0.0f;
    for (int i = t; i < LL; i += 256) s += p[i];
    __shared__ float red[256];
    red[t] = s; __syncthreads();
    for (int q = 128; q > 0; q >>= 1) { if (t < q) red[t] += red[t + q]; __syncthreads(); }
    if (t == 0) xgap[bc] = red[0] * (1.0f / (float)LL);
}

// W_eff[e][r][c] = in_w[e][r][c] * ln_g[e][c]  (f16)
__global__ void k_weff(const float* __restrict__ in_w, const float* __restrict__ ln_g,
                       h16* __restrict__ weff, int n)
{
    const int i = blockIdx.x * blockDim.x + threadIdx.x;
    if (i >= n) return;
    const int c = i & 127;
    const int e = i / (2 * DIN * CCH);
    weff[i] = (h16)(in_w[i] * ln_g[e * CCH + c]);
}

// b_eff[e][r] = dot(ln_b[e,:], in_w[e,r,:])  (f32).  Block per (e,r).
__global__ __launch_bounds__(128)
void k_beff(const float* __restrict__ in_w, const float* __restrict__ ln_b,
            float* __restrict__ beff)
{
    const int er = blockIdx.x;            // e*512 + r
    const int e  = er >> 9;
    const int c  = threadIdx.x;
    __shared__ float red[128];
    red[c] = ln_b[e * CCH + c] * in_w[(size_t)er * CCH + c];
    __syncthreads();
    for (int s = 64; s > 0; s >>= 1) { if (c < s) red[c] += red[c + s]; __syncthreads(); }
    if (c == 0) beff[er] = red[0];
}

__global__ void k_f2h(const float* __restrict__ a, h16* __restrict__ o, int n)
{
    const int i = blockIdx.x * blockDim.x + threadIdx.x;
    if (i < n) o[i] = (h16)a[i];
}

// Depthwise causal conv (4 taps) + bias + SiLU.  xm lives in xz[..., 0:256].
__global__ void k_conv(const float* __restrict__ xz, const float* __restrict__ cw,
                       const float* __restrict__ cb, float* __restrict__ xc,
                       h16* __restrict__ xch, int n)
{
    const int i = blockIdx.x * blockDim.x + threadIdx.x;   // over E*B*L*DIN
    if (i >= n) return;
    const int d  = i & 255;
    const int l  = (i >> 8) & 1023;
    const int eb = i >> 18;                                // e*B + b
    const int e  = eb >> 2;
    const float* base = xz + ((size_t)eb * LL) * 512 + d;
    const float* w = cw + ((size_t)e * DIN + d) * DCONV;
    float acc = cb[e * DIN + d];
    #pragma unroll
    for (int k = 0; k < DCONV; ++k) {
        const int l0 = l + k - (DCONV - 1);
        const float xv = (l0 >= 0) ? base[(size_t)l0 * 512] : 0.0f;
        acc += w[k] * xv;
    }
    const float s = acc / (1.0f + __expf(-acc));
    xc[i]  = s;
    xch[i] = (h16)s;
}

// dt = softplus(dt_r @ dtproj_w^T + dtproj_b);  dt_r = dbc[..., 0:8] (stride 48)
__global__ void k_dt(const float* __restrict__ dbc, const float* __restrict__ dtw,
                     const float* __restrict__ dtb, float* __restrict__ dt, int n)
{
    const int i = blockIdx.x * blockDim.x + threadIdx.x;   // over E*B*L*DIN
    if (i >= n) return;
    const int d   = i & 255;
    const int ebl = i >> 8;
    const int e   = ebl >> 12;                             // B*L = 4096
    const float* r = dbc + (size_t)ebl * 48;
    const float* w = dtw + ((size_t)e * DIN + d) * DTRANK;
    float acc = dtb[e * DIN + d];
    #pragma unroll
    for (int k = 0; k < DTRANK; ++k) acc += r[k] * w[k];
    dt[i] = (acc > 20.0f) ? acc : log1pf(__expf(acc));
}

// Selective scan: one block per (e,b); thread = channel d; 16 states in regs.
// The 1024-step serial chain is the latency bottleneck of the whole model, so
// we issue global_prefetch (CDNA5 GLOBAL_PREFETCH_B8) a few steps ahead to
// keep the next rows of dt/u/z resident while the chain executes.
__global__ __launch_bounds__(256)
void k_scan(const float* __restrict__ dbc, const float* __restrict__ dt,
            const float* __restrict__ xc, const float* __restrict__ xz,
            const float* __restrict__ A_log, const float* __restrict__ Dp,
            h16* __restrict__ yh)
{
    const int eb = blockIdx.x;         // e*B + b
    const int e  = eb >> 2;
    const int d  = threadIdx.x;
    float Arow[DSTATE];
    #pragma unroll
    for (int s = 0; s < DSTATE; ++s)
        Arow[s] = -__expf(A_log[((size_t)e * DIN + d) * DSTATE + s]);
    const float Dd = Dp[e * DIN + d];
    float h[DSTATE];
    #pragma unroll
    for (int s = 0; s < DSTATE; ++s) h[s] = 0.0f;

    __shared__ float Bs[DSTATE];
    __shared__ float Cs[DSTATE];
    const size_t baseDbc = (size_t)eb * LL * 48;
    const size_t baseD   = (size_t)eb * LL * DIN;
    const size_t baseZ   = (size_t)eb * LL * 512;

    #define PFD 8
    for (int l = 0; l < LL; ++l) {
        if (l + PFD < LL) {
            __builtin_prefetch(&dt[baseD + (size_t)(l + PFD) * DIN + d], 0, 1);
            __builtin_prefetch(&xc[baseD + (size_t)(l + PFD) * DIN + d], 0, 1);
            __builtin_prefetch(&xz[baseZ + (size_t)(l + PFD) * 512 + 256 + d], 0, 1);
        }
        __syncthreads();
        if (d < DSTATE)             Bs[d]          = dbc[baseDbc + (size_t)l * 48 + 8  + d];
        else if (d < 2 * DSTATE)    Cs[d - DSTATE] = dbc[baseDbc + (size_t)l * 48 + 24 + (d - DSTATE)];
        __syncthreads();
        const float dtv = dt[baseD + (size_t)l * DIN + d];
        const float u   = xc[baseD + (size_t)l * DIN + d];
        const float du  = dtv * u;
        float y = 0.0f;
        #pragma unroll
        for (int s = 0; s < DSTATE; ++s) {
            h[s] = h[s] * __expf(dtv * Arow[s]) + du * Bs[s];
            y += h[s] * Cs[s];
        }
        const float z = xz[baseZ + (size_t)l * 512 + 256 + d];
        const float o = (y + Dd * u) * (z / (1.0f + __expf(-z)));
        yh[baseD + (size_t)l * DIN + d] = (h16)o;
    }
    #undef PFD
}

// Gating: softmax over E=4, top-2 (first-occurrence ties like jax), aux loss.
__global__ __launch_bounds__(32)
void k_gate(const float* __restrict__ xgap, const float* __restrict__ gates,
            float* __restrict__ Wm, float* __restrict__ loss_out)
{
    const int t = threadIdx.x;
    __shared__ float probs[4][4][4];   // [g][b][e]
    if (t < 16) {
        const int g = t >> 2, b = t & 3;
        float logit[4];
        for (int e2 = 0; e2 < 4; ++e2) {
            float s = 0.0f;
            for (int c = 0; c < CCH; ++c)
                s += xgap[b * CCH + c] * gates[((size_t)g * CCH + c) * EXPERTS + e2];
            logit[e2] = s;
        }
        const float mx = fmaxf(fmaxf(logit[0], logit[1]), fmaxf(logit[2], logit[3]));
        float p[4], den = 0.0f;
        for (int e2 = 0; e2 < 4; ++e2) { p[e2] = __expf(logit[e2] - mx); den += p[e2]; }
        for (int e2 = 0; e2 < 4; ++e2) { p[e2] /= den; probs[g][b][e2] = p[e2]; }
        int i0 = 0;
        for (int e2 = 1; e2 < 4; ++e2) if (p[e2] > p[i0]) i0 = e2;
        int i1 = -1;
        for (int e2 = 0; e2 < 4; ++e2) {
            if (e2 == i0) continue;
            if (i1 < 0 || p[e2] > p[i1]) i1 = e2;
        }
        const float s2 = p[i0] + p[i1] + 1e-10f;
        for (int e2 = 0; e2 < 4; ++e2) {
            const float wv = (e2 == i0) ? p[i0] / s2 : (e2 == i1) ? p[i1] / s2 : 0.0f;
            Wm[(g * BB + b) * EXPERTS + e2] = wv;
        }
    }
    __syncthreads();
    if (t == 0) {
        float loss = 0.0f;
        for (int g = 0; g < 4; ++g) {
            float usage[4], m = 0.0f;
            for (int e2 = 0; e2 < 4; ++e2) {
                float u = 0.0f;
                for (int b = 0; b < 4; ++b) u += probs[g][b][e2];
                u *= 0.25f; usage[e2] = u; m += u;
            }
            m *= 0.25f;
            float var = 0.0f;
            for (int e2 = 0; e2 < 4; ++e2) { const float dd = usage[e2] - m; var += dd * dd; }
            var *= (1.0f / 3.0f);          // ddof = 1
            loss += var / (m * m + 1e-10f);
        }
        *loss_out = loss;
    }
}

// outs[g][b][c][h][w] = sum_e Wm[g,b,e] * eo[e,b,l,c]
__global__ void k_combine(const float* __restrict__ eo, const float* __restrict__ Wm,
                          float* __restrict__ out, int n)
{
    const int i = blockIdx.x * blockDim.x + threadIdx.x;   // over 4*B*C*HW
    if (i >= n) return;
    const int l  = i & 1023;
    const int c  = (i >> 10) & 127;
    const int gb = i >> 17;            // g*B + b
    const int b  = gb & 3, g = gb >> 2;
    float acc = 0.0f;
    #pragma unroll
    for (int e2 = 0; e2 < 4; ++e2)
        acc += Wm[(g * BB + b) * EXPERTS + e2]
             * eo[(((size_t)e2 * BB + b) * LL + l) * CCH + c];
    out[i] = acc;
}

// ---------------------------------------------------------------------------
extern "C" void kernel_launch(void* const* d_in, const int* in_sizes, int n_in,
                              void* d_out, int out_size, void* d_ws, size_t ws_size,
                              hipStream_t stream)
{
    const float* x        = (const float*)d_in[0];
    const float* gates    = (const float*)d_in[1];
    const float* ln_g     = (const float*)d_in[2];
    const float* ln_b     = (const float*)d_in[3];
    const float* in_w     = (const float*)d_in[4];
    const float* conv_w   = (const float*)d_in[5];
    const float* conv_b   = (const float*)d_in[6];
    const float* xproj_w  = (const float*)d_in[7];
    const float* dtproj_w = (const float*)d_in[8];
    const float* dtproj_b = (const float*)d_in[9];
    const float* A_log    = (const float*)d_in[10];
    const float* Dp       = (const float*)d_in[11];
    const float* out_w    = (const float*)d_in[12];
    const float* proj_w   = (const float*)d_in[13];
    const float* proj_b   = (const float*)d_in[14];
    float* out = (float*)d_out;
    (void)in_sizes; (void)n_in; (void)out_size; (void)ws_size;

    char* p = (char*)d_ws;
    auto alloc = [&](size_t bytes) -> char* {
        char* r = p; p += (bytes + 255) & ~(size_t)255; return r;
    };
    h16*   xhat_h = (h16*)  alloc((size_t)MROWS * CCH * 2);
    h16*   weff_h = (h16*)  alloc((size_t)EXPERTS * 2 * DIN * CCH * 2);
    float* beff   = (float*)alloc((size_t)EXPERTS * 2 * DIN * 4);
    h16*   xpw_h  = (h16*)  alloc((size_t)EXPERTS * 40 * DIN * 2);
    h16*   ow_h   = (h16*)  alloc((size_t)EXPERTS * CCH * DIN * 2);
    h16*   pw_h   = (h16*)  alloc((size_t)EXPERTS * CCH * CCH * 2);
    float* xgap   = (float*)alloc((size_t)BB * CCH * 4);
    float* Wm     = (float*)alloc((size_t)4 * BB * EXPERTS * 4);
    float* xz     = (float*)alloc((size_t)EXPERTS * MROWS * 512 * 4);
    float* xc     = (float*)alloc((size_t)EXPERTS * MROWS * DIN * 4);
    h16*   xc_h   = (h16*)  alloc((size_t)EXPERTS * MROWS * DIN * 2);
    float* dbc    = (float*)alloc((size_t)EXPERTS * MROWS * 48 * 4);
    float* dtv    = (float*)alloc((size_t)EXPERTS * MROWS * DIN * 4);
    h16*   y_h    = (h16*)  alloc((size_t)EXPERTS * MROWS * DIN * 2);
    h16*   o1_h   = (h16*)  alloc((size_t)EXPERTS * MROWS * CCH * 2);
    float* eo     = (float*)alloc((size_t)EXPERTS * MROWS * CCH * 4);

    // Stage 0: LN / GAP / weight prep (fold LN affine into in_proj)
    k_ln <<<MROWS, 128, 0, stream>>>(x, xhat_h);
    k_gap<<<BB * CCH, 256, 0, stream>>>(x, xgap);
    { int n = EXPERTS * 2 * DIN * CCH; k_weff<<<(n + 255) / 256, 256, 0, stream>>>(in_w, ln_g, weff_h, n); }
    k_beff<<<EXPERTS * 2 * DIN, 128, 0, stream>>>(in_w, ln_b, beff);
    { int n = EXPERTS * 40 * DIN;  k_f2h<<<(n + 255) / 256, 256, 0, stream>>>(xproj_w, xpw_h, n); }
    { int n = EXPERTS * CCH * DIN; k_f2h<<<(n + 255) / 256, 256, 0, stream>>>(out_w, ow_h, n); }
    { int n = EXPERTS * CCH * CCH; k_f2h<<<(n + 255) / 256, 256, 0, stream>>>(proj_w, pw_h, n); }

    // GEMM1: xz = xhat @ W_eff^T + b_eff   (M=4096, K=128, N=512)
    {
        dim3 grid(4, MROWS / 16, EXPERTS);
        k_gemm_f16<false><<<grid, 256, 0, stream>>>(xhat_h, CCH, 0,
                                             weff_h, CCH, (long)2 * DIN * CCH,
                                             beff, 2 * DIN,
                                             xz, 512, (long)MROWS * 512, 0,
                                             CCH, 512, 32);
    }
    { int n = EXPERTS * MROWS * DIN; k_conv<<<(n + 255) / 256, 256, 0, stream>>>(xz, conv_w, conv_b, xc, xc_h, n); }

    // GEMM2: dbc = xc @ xproj_w^T   (M=4096, K=256, Nvalid=40, padded to 48)
    {
        dim3 grid(1, MROWS / 16, EXPERTS);
        k_gemm_f16<true><<<grid, 256, 0, stream>>>(xc_h, DIN, (long)MROWS * DIN,
                                             xpw_h, DIN, (long)40 * DIN,
                                             nullptr, 0,
                                             dbc, 48, (long)MROWS * 48, 0,
                                             DIN, 40, 3);
    }
    { int n = EXPERTS * MROWS * DIN; k_dt<<<(n + 255) / 256, 256, 0, stream>>>(dbc, dtproj_w, dtproj_b, dtv, n); }

    k_scan<<<EXPERTS * BB, DIN, 0, stream>>>(dbc, dtv, xc, xz, A_log, Dp, y_h);

    // GEMM4: o1 = y @ out_w^T   (K=256, N=128, f16 out)
    {
        dim3 grid(1, MROWS / 16, EXPERTS);
        k_gemm_f16<false><<<grid, 256, 0, stream>>>(y_h, DIN, (long)MROWS * DIN,
                                             ow_h, DIN, (long)CCH * DIN,
                                             nullptr, 0,
                                             o1_h, CCH, (long)MROWS * CCH, 1,
                                             DIN, CCH, 8);
    }
    // GEMM5: eo = o1 @ proj_w^T + proj_b   (K=128, N=128, f32 out)
    {
        dim3 grid(1, MROWS / 16, EXPERTS);
        k_gemm_f16<false><<<grid, 256, 0, stream>>>(o1_h, CCH, (long)MROWS * CCH,
                                             pw_h, CCH, (long)CCH * CCH,
                                             proj_b, CCH,
                                             eo, CCH, (long)MROWS * CCH, 0,
                                             CCH, CCH, 8);
    }

    k_gate<<<1, 32, 0, stream>>>(xgap, gates, Wm, out + (size_t)4 * BB * CCH * LL);
    { int n = 4 * BB * CCH * LL; k_combine<<<(n + 255) / 256, 256, 0, stream>>>(eo, Wm, out, n); }
}
